// MambaBlock_81999515615487
// MI455X (gfx1250) — compile-verified, matrix operand order checked by reading, not run on hardware
//
#include <hip/hip_runtime.h>
#include <math.h>

// MI455X / gfx1250 implementation of the Mamba image block.
// All dense projections run on v_wmma_f32_16x16x32_f16 (fp16 in, fp32 acc),
// K-loop fully unrolled via template, 4 M-tiles per wave per B-fragment.
// Requires ws_size >= ~99 MB (24.7M floats).

typedef __attribute__((ext_vector_type(16))) _Float16 v16h;
typedef __attribute__((ext_vector_type(8)))  float    v8f;
typedef _Float16 h16;

#define BSZ 2
#define CD  48      // d_model
#define DI  96      // expanded dim
#define HWP 16384   // 128*128
#define IMG 128
#define L1  16384
#define L2T 8192    // layer2 sequence length (2*64*64)

__device__ __forceinline__ float sigm_(float x) {
  return __builtin_amdgcn_rcpf(1.0f + __expf(-x));   // v_rcp_f32, no div expansion
}

// ======================= WMMA GEMM =======================
// C[M,Npad] = A[M,KPAD](f16) * Bw[KPAD,Npad](f16) + bias ; ACT: 0 none, 1 sigmoid
// One wave per 64x16 output tile (4 stacked 16x16 WMMA tiles share each B frag).
// KPAD % 32 == 0, M % 64 == 0, Npad % 16 == 0.
template<int KPAD, int ACT>
__global__ void k_wmma_gemm(const h16* __restrict__ A, const h16* __restrict__ Bw,
                            const float* __restrict__ bias, float* __restrict__ C,
                            int M, int Npad)
{
  const int m0   = blockIdx.x * 64;
  const int n0   = blockIdx.y * 16;
  const int lane = threadIdx.x & 31;
  const int hsel = lane >> 4;        // which K-half this lane holds
  const int mn   = lane & 15;        // M index (A) / N index (B,D)
  const int kh   = hsel * 8;
  v8f acc[4] = {};
#pragma unroll
  for (int k0 = 0; k0 < KPAD; k0 += 32) {
    v16h bf;
    const h16* bc = Bw + (size_t)k0 * Npad + (n0 + mn);
#pragma unroll
    for (int p = 0; p < 8; ++p) {
      int kk = ((p < 4) ? (2 * p) : (16 + 2 * (p - 4))) + kh;
      bf[2*p]   = bc[(size_t)kk * Npad];
      bf[2*p+1] = bc[(size_t)(kk + 1) * Npad];
    }
#pragma unroll
    for (int t = 0; t < 4; ++t) {
      v16h af;
      const h16* ar = A + (size_t)(m0 + t * 16 + mn) * KPAD + k0;
#pragma unroll
      for (int p = 0; p < 8; ++p) {
        int kk = ((p < 4) ? (2 * p) : (16 + 2 * (p - 4))) + kh;
        af[2*p]   = ar[kk];
        af[2*p+1] = ar[kk + 1];
      }
      acc[t] = __builtin_amdgcn_wmma_f32_16x16x32_f16(false, af, false, bf,
                                                      (short)0, acc[t], false, false);
    }
  }
  const int n = n0 + mn;
  const float bv = bias ? bias[n] : 0.0f;
#pragma unroll
  for (int t = 0; t < 4; ++t) {
#pragma unroll
    for (int r = 0; r < 8; ++r) {
      int m = m0 + t * 16 + r + hsel * 8;
      float v = acc[t][r] + bv;
      if (ACT == 1) v = sigm_(v);
      C[(size_t)m * Npad + n] = v;
    }
  }
}

// Wt[Kpad,Npad](f16) = transpose(W[N,K]) zero-padded
__global__ void k_prep_w(const float* __restrict__ Wsrc, h16* __restrict__ Wt,
                         int N, int K, int Npad, int Kpad)
{
  int i = blockIdx.x * blockDim.x + threadIdx.x;
  int tot = Kpad * Npad;
  if (i >= tot) return;
  int k = i / Npad, n = i % Npad;
  Wt[i] = (h16)((n < N && k < K) ? Wsrc[(size_t)n * K + k] : 0.0f);
}

// ======================= elementwise / layout =======================
__global__ void k_copy(const float* __restrict__ a, float* __restrict__ b, int n) {
  int i = blockIdx.x * blockDim.x + threadIdx.x;
  if (i < n) b[i] = a[i];
}

// LayerNorm over channel axis of (B,C=48,HW) channel-major tensor
__global__ void k_ln_chan(const float* __restrict__ in, const float* __restrict__ w,
                          const float* __restrict__ bb, float* __restrict__ out)
{
  int t = blockIdx.x * blockDim.x + threadIdx.x;
  if (t >= BSZ * HWP) return;
  int b = t / HWP, p = t % HWP;
  const float* src = in + (size_t)b * CD * HWP + p;
  float v[CD]; float mu = 0.f;
#pragma unroll
  for (int c = 0; c < CD; ++c) { v[c] = src[(size_t)c * HWP]; mu += v[c]; }
  mu *= (1.0f / CD);
  float var = 0.f;
#pragma unroll
  for (int c = 0; c < CD; ++c) { float d = v[c] - mu; var += d * d; }
  float rs = rsqrtf(var * (1.0f / CD) + 1e-5f);
  float* dst = out + (size_t)b * CD * HWP + p;
#pragma unroll
  for (int c = 0; c < CD; ++c) dst[(size_t)c * HWP] = (v[c] - mu) * rs * w[c] + bb[c];
}

// LN over channels, emitted token-major fp16 padded to 64 (A-matrix for FFN pi GEMM)
__global__ void k_ln_tok_h(const float* __restrict__ in, const float* __restrict__ w,
                           const float* __restrict__ bb, h16* __restrict__ Ah)
{
  int t = blockIdx.x * blockDim.x + threadIdx.x;
  if (t >= BSZ * HWP) return;
  int b = t / HWP, p = t % HWP;
  const float* src = in + (size_t)b * CD * HWP + p;
  float v[CD]; float mu = 0.f;
#pragma unroll
  for (int c = 0; c < CD; ++c) { v[c] = src[(size_t)c * HWP]; mu += v[c]; }
  mu *= (1.0f / CD);
  float var = 0.f;
#pragma unroll
  for (int c = 0; c < CD; ++c) { float d = v[c] - mu; var += d * d; }
  float rs = rsqrtf(var * (1.0f / CD) + 1e-5f);
  h16* dst = Ah + (size_t)t * 64;
#pragma unroll
  for (int c = 0; c < CD; ++c) dst[c] = (h16)((v[c] - mu) * rs * w[c] + bb[c]);
#pragma unroll
  for (int c = CD; c < 64; ++c) dst[c] = (h16)0.0f;
}

// layer1 horizontal ordering: l = (hg*64+wg)*4 + i*2 + j -> (h=2hg+i, w=2wg+j)
__global__ void k_reorder_h(const float* __restrict__ xln, h16* __restrict__ u) {
  int t = blockIdx.x * blockDim.x + threadIdx.x;
  if (t >= BSZ * L1) return;
  int b = t / L1, l = t % L1;
  int j = l & 1, i = (l >> 1) & 1, wg = (l >> 2) & 63, hg = l >> 8;
  int p = (2 * hg + i) * IMG + (2 * wg + j);
  const float* src = xln + (size_t)b * CD * HWP + p;
  h16* dst = u + (size_t)t * 64;
#pragma unroll
  for (int c = 0; c < CD; ++c) dst[c] = (h16)src[(size_t)c * HWP];
#pragma unroll
  for (int c = CD; c < 64; ++c) dst[c] = (h16)0.0f;
}
__global__ void k_scatter_h(const float* __restrict__ y, float* __restrict__ out) {
  int idx = blockIdx.x * blockDim.x + threadIdx.x;
  if (idx >= BSZ * L1 * CD) return;
  int c = idx % CD; int t = idx / CD; int b = t / L1, l = t % L1;
  int j = l & 1, i = (l >> 1) & 1, wg = (l >> 2) & 63, hg = l >> 8;
  int p = (2 * hg + i) * IMG + (2 * wg + j);
  out[((size_t)b * CD + c) * HWP + p] += y[(size_t)t * CD + c];
}
// layer1 vertical ordering: l = (wg*64+hg)*4 + j*2 + i
__global__ void k_reorder_v(const float* __restrict__ xln, h16* __restrict__ u) {
  int t = blockIdx.x * blockDim.x + threadIdx.x;
  if (t >= BSZ * L1) return;
  int b = t / L1, l = t % L1;
  int i = l & 1, j = (l >> 1) & 1, hg = (l >> 2) & 63, wg = l >> 8;
  int p = (2 * hg + i) * IMG + (2 * wg + j);
  const float* src = xln + (size_t)b * CD * HWP + p;
  h16* dst = u + (size_t)t * 64;
#pragma unroll
  for (int c = 0; c < CD; ++c) dst[c] = (h16)src[(size_t)c * HWP];
#pragma unroll
  for (int c = CD; c < 64; ++c) dst[c] = (h16)0.0f;
}
__global__ void k_scatter_v(const float* __restrict__ y, float* __restrict__ out) {
  int idx = blockIdx.x * blockDim.x + threadIdx.x;
  if (idx >= BSZ * L1 * CD) return;
  int c = idx % CD; int t = idx / CD; int b = t / L1, l = t % L1;
  int i = l & 1, j = (l >> 1) & 1, hg = (l >> 2) & 63, wg = l >> 8;
  int p = (2 * hg + i) * IMG + (2 * wg + j);
  out[((size_t)b * CD + c) * HWP + p] += y[(size_t)t * CD + c];
}

// ======================= mamba internals =======================
// causal depthwise conv1d (D_CONV=4) on x-half of xz, + SiLU; emits f32 + f16
__global__ void k_conv_silu(const float* __restrict__ xz, const float* __restrict__ cw,
                            const float* __restrict__ cb, float* __restrict__ xc,
                            h16* __restrict__ xch, int Ln)
{
  int idx = blockIdx.x * blockDim.x + threadIdx.x;
  if (idx >= BSZ * Ln * DI) return;
  int d = idx % DI; int m = idx / DI; int b = m / Ln, l = m % Ln;
  float s = cb[d];
#pragma unroll
  for (int k = 0; k < 4; ++k) {
    int lp = l - 3 + k;
    if (lp >= 0) s += cw[k * DI + d] * xz[((size_t)(b * Ln + lp)) * 192 + d];
  }
  float v = s * sigm_(s);
  xc[(size_t)m * DI + d] = v;
  xch[(size_t)m * DI + d] = (h16)v;
}

// delta = softplus(dtr @ dtproj_w.T + b), dtr = xdbl[:, :3]
__global__ void k_dtproj(const float* __restrict__ xdbl, const float* __restrict__ dtw,
                         const float* __restrict__ dtb, float* __restrict__ delt, int M)
{
  int idx = blockIdx.x * blockDim.x + threadIdx.x;
  if (idx >= M * DI) return;
  int d = idx % DI; int m = idx / DI;
  float a = dtb[d];
#pragma unroll
  for (int r = 0; r < 3; ++r) a += xdbl[(size_t)m * 48 + r] * dtw[d * 3 + r];
  delt[(size_t)m * DI + d] = (a > 20.0f) ? a : log1pf(__expf(a));
}

// selective scan: 16 lanes = 16 states per (b,d) channel. 192 channels total.
// y = (sum_s h*C + u*D) * silu(z), stored fp16 (A-matrix for out-proj GEMM)
// Latency-bound (16384 dependent steps): prefetch t+16's rows while the
// recurrence chain serializes (global_prefetch_b8 on gfx1250).
__global__ void k_scan(const float* __restrict__ xc, const float* __restrict__ delt,
                       const float* __restrict__ xdbl, const float* __restrict__ xz,
                       const float* __restrict__ A_log, const float* __restrict__ Dv,
                       h16* __restrict__ yh, int Ln)
{
  int gid = (blockIdx.x * blockDim.x + threadIdx.x) >> 4;
  int s = threadIdx.x & 15;
  if (gid >= BSZ * DI) return;
  int b = gid / DI, d = gid % DI;
  float Ad = -__expf(A_log[d * 16 + s]);
  float Dd = Dv[d];
  float h = 0.0f;
  size_t base = (size_t)b * Ln;
  for (int t = 0; t < Ln; ++t) {
    size_t m = base + t;
    if (t + 16 < Ln) {
      size_t mp = m + 16;
      __builtin_prefetch(&delt[mp * DI + d], 0, 0);
      __builtin_prefetch(&xc[mp * DI + d], 0, 0);
      __builtin_prefetch(&xdbl[mp * 48 + 3 + s], 0, 0);
    }
    float dt = delt[m * DI + d];
    float u  = xc[m * DI + d];
    float Bt = xdbl[m * 48 + 3 + s];
    float Ct = xdbl[m * 48 + 19 + s];
    h = h * __expf(dt * Ad) + (dt * u) * Bt;
    float part = h * Ct;
    part += __shfl_xor(part, 8, 32);
    part += __shfl_xor(part, 4, 32);
    part += __shfl_xor(part, 2, 32);
    part += __shfl_xor(part, 1, 32);
    if (s == 0) {
      float z = xz[m * 192 + 96 + d];
      float y = (part + u * Dd) * (z * sigm_(z));
      yh[m * DI + d] = (h16)y;
    }
  }
}

// ======================= layer 2 =======================
__global__ void k_pool2(const float* __restrict__ xln, float* __restrict__ x2) {
  int idx = blockIdx.x * blockDim.x + threadIdx.x;
  if (idx >= BSZ * CD * 4096) return;
  int p2 = idx % 4096; int bc = idx / 4096;
  int h2 = p2 >> 6, w2 = p2 & 63;
  const float* s = xln + (size_t)bc * HWP;
  float v = s[(2*h2)*IMG + 2*w2] + s[(2*h2)*IMG + 2*w2+1]
          + s[(2*h2+1)*IMG + 2*w2] + s[(2*h2+1)*IMG + 2*w2+1];
  x2[idx] = 0.25f * v;
}
// sequence = concat(flip(rowmajor), flip(colmajor)) of pooled 64x64
__global__ void k_reorder2(const float* __restrict__ x2, h16* __restrict__ u) {
  int t = blockIdx.x * blockDim.x + threadIdx.x;
  if (t >= BSZ * L2T) return;
  int b = t / L2T, l = t % L2T;
  int sp;
  if (l < 4096) sp = 4095 - l;
  else { int idx = 8191 - l; sp = (idx & 63) * 64 + (idx >> 6); }
  const float* src = x2 + (size_t)b * CD * 4096 + sp;
  h16* dst = u + (size_t)t * 64;
#pragma unroll
  for (int c = 0; c < CD; ++c) dst[c] = (h16)src[(size_t)c * 4096];
#pragma unroll
  for (int c = CD; c < 64; ++c) dst[c] = (h16)0.0f;
}
// ysmall[b,c,h2,w2] = y2[4095-(h2*64+w2)] + y2[8191-(w2*64+h2)]
__global__ void k_unscatter2(const float* __restrict__ y2, float* __restrict__ ysm) {
  int idx = blockIdx.x * blockDim.x + threadIdx.x;
  if (idx >= BSZ * CD * 4096) return;
  int p2 = idx % 4096; int bc = idx / 4096; int b = bc / CD, c = bc % CD;
  int h2 = p2 >> 6, w2 = p2 & 63;
  size_t r0 = (size_t)b * L2T + (4095 - (h2 * 64 + w2));
  size_t r1 = (size_t)b * L2T + (8191 - (w2 * 64 + h2));
  ysm[idx] = y2[r0 * CD + c] + y2[r1 * CD + c];
}
// bilinear x2 upsample (half-pixel centers), accumulate into out
__global__ void k_upsample_add(const float* __restrict__ ysm, float* __restrict__ out) {
  int idx = blockIdx.x * blockDim.x + threadIdx.x;
  if (idx >= BSZ * CD * HWP) return;
  int p = idx % HWP; int bc = idx / HWP;
  int h = p / IMG, w = p % IMG;
  float fy = fminf(fmaxf(0.5f * h - 0.25f, 0.0f), 63.0f);
  float fx = fminf(fmaxf(0.5f * w - 0.25f, 0.0f), 63.0f);
  int y0 = (int)fy, x0 = (int)fx;
  int y1 = min(y0 + 1, 63), x1 = min(x0 + 1, 63);
  float wy = fy - y0, wx = fx - x0;
  const float* s = ysm + (size_t)bc * 4096;
  float v = (1-wy)*((1-wx)*s[y0*64+x0] + wx*s[y0*64+x1])
          +    wy *((1-wx)*s[y1*64+x0] + wx*s[y1*64+x1]);
  out[idx] += v;
}

// ======================= FFN =======================
// grouped 3x3 conv (96 out ch, 2 in ch per group of 48), token-major xh (M,96)
__global__ void k_dwconv(const float* __restrict__ xh, const float* __restrict__ dww,
                         const float* __restrict__ dwb, float* __restrict__ loc)
{
  int idx = blockIdx.x * blockDim.x + threadIdx.x;
  if (idx >= BSZ * HWP * DI) return;
  int o = idx % DI; int m = idx / DI; int b = m / HWP, p = m % HWP;
  int h = p / IMG, w = p % IMG, g = o >> 1;
  float s = dwb[o];
  for (int dy = -1; dy <= 1; ++dy) {
    int hh = h + dy; if (hh < 0 || hh >= IMG) continue;
    for (int dx = -1; dx <= 1; ++dx) {
      int ww = w + dx; if (ww < 0 || ww >= IMG) continue;
      const float* src = xh + ((size_t)b * HWP + hh * IMG + ww) * DI + g * 2;
      const float* kw = dww + ((o * 2) * 3 + (dy + 1)) * 3 + (dx + 1);
      s += kw[0] * src[0] + kw[9] * src[1];   // ci=0,1 ; stride over ci is 9
    }
  }
  loc[(size_t)m * DI + o] = s;
}
// LN over 96 channels (token-major); writes f32 + fused fp16 (cols 0..95 of 192)
__global__ void k_ln_mid_fused(const float* __restrict__ loc, const float* __restrict__ nw,
                               const float* __restrict__ nb, float* __restrict__ lln,
                               h16* __restrict__ fused, int M)
{
  int t = blockIdx.x * blockDim.x + threadIdx.x;
  if (t >= M) return;
  const float* src = loc + (size_t)t * DI;
  float mu = 0.f;
  for (int o = 0; o < DI; ++o) mu += src[o];
  mu *= (1.0f / DI);
  float var = 0.f;
  for (int o = 0; o < DI; ++o) { float d = src[o] - mu; var += d * d; }
  float rs = rsqrtf(var * (1.0f / DI) + 1e-5f);
  for (int o = 0; o < DI; ++o) {
    float v = (src[o] - mu) * rs * nw[o] + nb[o];
    lln[(size_t)t * DI + o] = v;
    fused[(size_t)t * 192 + o] = (h16)v;
  }
}
__global__ void k_gpool(const float* __restrict__ xh, float* __restrict__ g0) {
  int b = blockIdx.x / DI, o = blockIdx.x % DI;
  __shared__ float sm[256];
  float s = 0.f;
  for (int p = threadIdx.x; p < HWP; p += 256)
    s += xh[((size_t)b * HWP + p) * DI + o];
  sm[threadIdx.x] = s; __syncthreads();
  for (int off = 128; off; off >>= 1) {
    if ((int)threadIdx.x < off) sm[threadIdx.x] += sm[threadIdx.x + off];
    __syncthreads();
  }
  if (threadIdx.x == 0) g0[b * DI + o] = sm[0] * (1.0f / HWP);
}
__global__ void k_se(const float* __restrict__ g0, const float* __restrict__ w1,
                     const float* __restrict__ b1, const float* __restrict__ w2,
                     const float* __restrict__ b2, float* __restrict__ g2)
{
  __shared__ float g1[6];
  for (int b = 0; b < BSZ; ++b) {
    if (threadIdx.x < 6) {
      float a = b1[threadIdx.x];
      for (int c = 0; c < DI; ++c) a += w1[threadIdx.x * DI + c] * g0[b * DI + c];
      g1[threadIdx.x] = 0.5f * a * (1.0f + erff(a * 0.70710678118654752f)); // exact gelu
    }
    __syncthreads();
    if (threadIdx.x < DI) {
      float a = b2[threadIdx.x];
      for (int r = 0; r < 6; ++r) a += w2[threadIdx.x * 6 + r] * g1[r];
      g2[b * DI + threadIdx.x] = sigm_(a);
    }
    __syncthreads();
  }
}
__global__ void k_glob_fused(const float* __restrict__ xh, const float* __restrict__ g2,
                             float* __restrict__ glob, h16* __restrict__ fused)
{
  int idx = blockIdx.x * blockDim.x + threadIdx.x;
  if (idx >= BSZ * HWP * DI) return;
  int o = idx % DI; int m = idx / DI; int b = m / HWP;
  float v = g2[b * DI + o] * xh[(size_t)m * DI + o];
  glob[(size_t)m * DI + o] = v;
  fused[(size_t)m * 192 + DI + o] = (h16)v;
}
__global__ void k_mix(const float* __restrict__ lln, const float* __restrict__ glob,
                      const float* __restrict__ gw, h16* __restrict__ xmix, int M)
{
  int idx = blockIdx.x * blockDim.x + threadIdx.x;
  if (idx >= M * DI) return;
  float g = gw[idx];
  xmix[idx] = (h16)(lln[idx] * g + glob[idx] * (1.0f - g));
}
__global__ void k_add_out(const float* __restrict__ ffo, float* __restrict__ out) {
  int idx = blockIdx.x * blockDim.x + threadIdx.x;
  if (idx >= BSZ * HWP * CD) return;
  int c = idx % CD; int t = idx / CD; int b = t / HWP, p = t % HWP;
  out[((size_t)b * CD + c) * HWP + p] += ffo[(size_t)t * CD + c];
}

// ======================= host =======================
struct MP { const float *in_w,*conv_w,*conv_b,*xproj_w,*dtproj_w,*dtproj_b,*A_log,*Dv,*out_w; };

static inline dim3 NB(size_t n) { return dim3((unsigned)((n + 255) / 256)); }

static void run_mamba(const MP& p, h16* useq, float* xz, float* xc, h16* xch,
                      float* xdbl, float* delt, h16* yh, float* yseq,
                      h16* wt_in, h16* wt_xp, h16* wt_out, int Ln, hipStream_t s)
{
  int M = BSZ * Ln;
  k_prep_w<<<NB(64 * 192), 256, 0, s>>>(p.in_w,    wt_in,  192, 48, 192, 64);
  k_prep_w<<<NB(96 * 48),  256, 0, s>>>(p.xproj_w, wt_xp,   35, 96,  48, 96);
  k_prep_w<<<NB(96 * 48),  256, 0, s>>>(p.out_w,   wt_out,  48, 96,  48, 96);
  k_wmma_gemm<64, 0><<<dim3(M / 64, 12), 32, 0, s>>>(useq, wt_in, nullptr, xz, M, 192);
  k_conv_silu<<<NB((size_t)M * DI), 256, 0, s>>>(xz, p.conv_w, p.conv_b, xc, xch, Ln);
  k_wmma_gemm<96, 0><<<dim3(M / 64, 3), 32, 0, s>>>(xch, wt_xp, nullptr, xdbl, M, 48);
  k_dtproj<<<NB((size_t)M * DI), 256, 0, s>>>(xdbl, p.dtproj_w, p.dtproj_b, delt, M);
  k_scan<<<12, 256, 0, s>>>(xc, delt, xdbl, xz, p.A_log, p.Dv, yh, Ln);
  k_wmma_gemm<96, 0><<<dim3(M / 64, 3), 32, 0, s>>>(yh, wt_out, nullptr, yseq, M, 48);
}

extern "C" void kernel_launch(void* const* d_in, const int* in_sizes, int n_in,
                              void* d_out, int out_size, void* d_ws, size_t ws_size,
                              hipStream_t stream)
{
  (void)in_sizes; (void)n_in; (void)out_size; (void)ws_size;
  auto F = [&](int i) { return (const float*)d_in[i]; };
  const float* X = F(0);
  const float *n1w=F(1), *n1b=F(2), *n2w=F(3), *n2b=F(4), *n3w=F(5), *n3b=F(6);
  MP m1 = {F(7),F(8),F(9),F(10),F(11),F(12),F(13),F(14),F(15)};
  MP m2 = {F(16),F(17),F(18),F(19),F(20),F(21),F(22),F(23),F(24)};
  MP m3 = {F(25),F(26),F(27),F(28),F(29),F(30),F(31),F(32),F(33)};
  const float *pi_w=F(34), *pi_b=F(35), *dw_w=F(36), *dw_b=F(37), *nw=F(38), *nb=F(39),
              *cm1w=F(40), *cm1b=F(41), *cm2w=F(42), *cm2b=F(43), *g_w=F(44), *g_b=F(45),
              *po_w=F(46), *po_b=F(47);
  float* out = (float*)d_out;
  float* W = (float*)d_ws;
  hipStream_t s = stream;

  // ---- workspace layout (float units). Total need ~24.7M floats (~99 MB). ----
  constexpr size_t SZ_IMG = (size_t)BSZ * CD * HWP;     // 1,572,864
  constexpr size_t oXLN = 0;
  constexpr size_t oWT0 = oXLN + SZ_IMG;                //  9216 f (18432 h)
  constexpr size_t oWT1 = oWT0 + 9216;                  //  3072 f ( 6144 h)
  constexpr size_t oWT2 = oWT1 + 3072;                  //  2304 f ( 4608 h)
  constexpr size_t AR   = oWT2 + 2304;
  // mamba-stage arena (sized for M = 32768 tokens)
  constexpr size_t oUSEQ = AR;                  // (M*64)h = 1,048,576 f
  constexpr size_t oXZ   = oUSEQ + 1048576;     // 6,291,456 f
  constexpr size_t oXC   = oXZ   + 6291456;     // 3,145,728 f
  constexpr size_t oXCH  = oXC   + 3145728;     // (M*96)h = 1,572,864 f
  constexpr size_t oXDBL = oXCH  + 1572864;     // 1,572,864 f
  constexpr size_t oDELT = oXDBL + 1572864;     // 3,145,728 f
  constexpr size_t oYH   = oDELT + 3145728;     // 1,572,864 f
  constexpr size_t oYSEQ = oYH   + 1572864;     // 1,572,864 f
  constexpr size_t oX2   = oYSEQ + 1572864;     //   393,216 f
  constexpr size_t oYSM  = oX2   + 393216;      //   393,216 f
  // ffn arena (reuses same region)
  constexpr size_t oAH   = AR;                  // 1,048,576 f
  constexpr size_t oXH   = oAH  + 1048576;      // 3,145,728 f
  constexpr size_t oLOC  = oXH  + 3145728;
  constexpr size_t oLLN  = oLOC + 3145728;
  constexpr size_t oFUS  = oLLN + 3145728;      // (M*192)h = 3,145,728 f
  constexpr size_t oGLO  = oFUS + 3145728;
  constexpr size_t oGW   = oGLO + 3145728;
  constexpr size_t oXMI  = oGW  + 3145728;      // (M*96)h = 1,572,864 f
  constexpr size_t oFFO  = oXMI + 1572864;
  constexpr size_t oG0   = oFFO + 1572864;
  constexpr size_t oG2   = oG0  + 256;

  float* XLN = W + oXLN;
  h16 *WT0 = (h16*)(W + oWT0), *WT1 = (h16*)(W + oWT1), *WT2 = (h16*)(W + oWT2);
  h16*   USEQ = (h16*)(W + oUSEQ);
  float* XZ = W + oXZ;   float* XC = W + oXC;   h16* XCH = (h16*)(W + oXCH);
  float* XDBL = W + oXDBL; float* DELT = W + oDELT;
  h16* YH = (h16*)(W + oYH); float* YSEQ = W + oYSEQ;
  float* X2 = W + oX2;   float* YSM = W + oYSM;
  h16* AH = (h16*)(W + oAH); float* XH = W + oXH; float* LOC = W + oLOC;
  float* LLN = W + oLLN; h16* FUS = (h16*)(W + oFUS); float* GLO = W + oGLO;
  float* GWB = W + oGW;  h16* XMI = (h16*)(W + oXMI); float* FFO = W + oFFO;
  float* G0 = W + oG0;   float* G2 = W + oG2;

  const int Mffn = BSZ * HWP;

  // ===== residual init + layer 1 (dual-direction mamba) =====
  k_copy<<<NB(SZ_IMG), 256, 0, s>>>(X, out, (int)SZ_IMG);
  k_ln_chan<<<NB((size_t)BSZ * HWP), 256, 0, s>>>(X, n1w, n1b, XLN);
  k_reorder_h<<<NB((size_t)BSZ * L1), 256, 0, s>>>(XLN, USEQ);
  run_mamba(m1, USEQ, XZ, XC, XCH, XDBL, DELT, YH, YSEQ, WT0, WT1, WT2, L1, s);
  k_scatter_h<<<NB((size_t)BSZ * L1 * CD), 256, 0, s>>>(YSEQ, out);
  k_reorder_v<<<NB((size_t)BSZ * L1), 256, 0, s>>>(XLN, USEQ);
  run_mamba(m2, USEQ, XZ, XC, XCH, XDBL, DELT, YH, YSEQ, WT0, WT1, WT2, L1, s);
  k_scatter_v<<<NB((size_t)BSZ * L1 * CD), 256, 0, s>>>(YSEQ, out);

  // ===== layer 2 (pooled flipped h+v mamba, bilinear upsample) =====
  k_ln_chan<<<NB((size_t)BSZ * HWP), 256, 0, s>>>(out, n2w, n2b, XLN);
  k_pool2<<<NB((size_t)BSZ * CD * 4096), 256, 0, s>>>(XLN, X2);
  k_reorder2<<<NB((size_t)BSZ * L2T), 256, 0, s>>>(X2, USEQ);
  run_mamba(m3, USEQ, XZ, XC, XCH, XDBL, DELT, YH, YSEQ, WT0, WT1, WT2, L2T, s);
  k_unscatter2<<<NB((size_t)BSZ * CD * 4096), 256, 0, s>>>(YSEQ, YSM);
  k_upsample_add<<<NB(SZ_IMG), 256, 0, s>>>(YSM, out);

  // ===== FFN =====
  k_ln_tok_h<<<NB((size_t)BSZ * HWP), 256, 0, s>>>(out, n3w, n3b, AH);
  k_prep_w<<<NB(64 * 96),  256, 0, s>>>(pi_w, WT1, 96, 48, 96, 64);
  k_prep_w<<<NB(192 * 96), 256, 0, s>>>(g_w,  WT0, 96, 192, 96, 192);
  k_prep_w<<<NB(96 * 48),  256, 0, s>>>(po_w, WT2, 48, 96, 48, 96);
  k_wmma_gemm<64, 0><<<dim3(Mffn / 64, 6), 32, 0, s>>>(AH, WT1, pi_b, XH, Mffn, 96);
  k_dwconv<<<NB((size_t)Mffn * DI), 256, 0, s>>>(XH, dw_w, dw_b, LOC);
  k_ln_mid_fused<<<NB((size_t)Mffn), 256, 0, s>>>(LOC, nw, nb, LLN, FUS, Mffn);
  k_gpool<<<dim3(BSZ * DI), 256, 0, s>>>(XH, G0);
  k_se<<<1, 128, 0, s>>>(G0, cm1w, cm1b, cm2w, cm2b, G2);
  k_glob_fused<<<NB((size_t)Mffn * DI), 256, 0, s>>>(XH, G2, GLO, FUS);
  k_wmma_gemm<192, 1><<<dim3(Mffn / 64, 6), 32, 0, s>>>(FUS, WT0, g_b, GWB, Mffn, 96);
  k_mix<<<NB((size_t)Mffn * DI), 256, 0, s>>>(LLN, GLO, GWB, XMI, Mffn);
  k_wmma_gemm<96, 0><<<dim3(Mffn / 64, 3), 32, 0, s>>>(XMI, WT2, po_b, FFO, Mffn, 48);
  k_add_out<<<NB((size_t)Mffn * CD), 256, 0, s>>>(FFO, out);
}
